// Critic_62740882260222
// MI455X (gfx1250) — matrix operand order repaired and checked by reading, not compiled, not run on hardware
//
#include <hip/hip_runtime.h>
#include <hip/hip_bf16.h>
#include <math.h>

// ---------------- CDNA5 WMMA types ----------------
typedef __attribute__((ext_vector_type(16))) __bf16 v16bf;
typedef __attribute__((ext_vector_type(8)))  __bf16 v8bf;
typedef __attribute__((ext_vector_type(8)))  float  v8f;

__device__ __forceinline__ v8f wmma_bf16(v16bf a, v16bf b, v8f c) {
  // D = A(16x32 bf16) * B(32x16 bf16) + C(16x16 f32)
  return __builtin_amdgcn_wmma_f32_16x16x32_bf16(
      false, a, false, b, (short)0, c, false, false);
}
__device__ __forceinline__ v16bf cat8(v8bf lo, v8bf hi) {
  return __builtin_shufflevector(lo, hi, 0,1,2,3,4,5,6,7,8,9,10,11,12,13,14,15);
}
// WGP-scope prefetch (builtin only reaches SE/SYS scopes)
__device__ __forceinline__ void prefetch_wgp(const void* p) {
  asm volatile("global_prefetch_b8 %0, off" :: "v"(p));
}

// ---------------- problem constants ----------------
constexpr int Bsz = 128, Tsz = 64, Csz = 64, Wsz = 27, Hsz = 128;
constexpr int Mtot = Bsz * Wsz;            // 3456 = 54 tiles of 64 (exact)
constexpr int HW   = Hsz * Wsz;            // 3456

// bf16 weight panels, K padded to multiples of 32, K order: k = tap*Cin + i
constexpr int KP_G0 = 576;   // 3*192
constexpr int KP_C0 = 608;   // 3*200=600 -> 608
constexpr int KP_G1 = 768;   // 3*256
constexpr int KP_C1 = 800;   // 3*264=792 -> 800
constexpr size_t W_G0 = 0;
constexpr size_t W_C0 = W_G0 + (size_t)256 * KP_G0;
constexpr size_t W_G1 = W_C0 + (size_t)128 * KP_C0;
constexpr size_t W_C1 = W_G1 + (size_t)256 * KP_G1;
constexpr size_t W_TOT = W_C1 + (size_t)128 * KP_C1;  // 524288 bf16

// workspace layout (bytes); activations channel-last [b][w][ch]
constexpr size_t SZ_HB  = (size_t)Bsz * HW * 2;       // bf16 hidden buf
constexpr size_t SZ_F   = (size_t)Bsz * HW * 4;       // f32 buf
constexpr size_t OFF_H0 = 0;
constexpr size_t OFF_H1 = OFF_H0 + SZ_HB;
constexpr size_t OFF_RH = OFF_H1 + SZ_HB;
constexpr size_t OFF_U  = OFF_RH + SZ_HB;
constexpr size_t OFF_PRE= OFF_U  + SZ_F;
constexpr size_t OFF_ST = OFF_PRE + SZ_F;             // 64 steps * 2 layers * 256 f32
constexpr size_t SZ_ST  = (size_t)Tsz * 2 * 256 * 4;  // 131072
constexpr size_t OFF_RW = OFF_ST + SZ_ST;             // 2 * 128 f32
constexpr size_t OFF_ZP = OFF_RW + 1024;              // 64B zero page for masked loads
constexpr size_t OFF_WT = OFF_ZP + 64;
constexpr size_t OFF_XIN= OFF_WT + W_TOT * 2;         // input repacked [t][b][w][c] bf16
constexpr size_t WS_NEED= OFF_XIN + (size_t)Bsz*Tsz*Csz*Wsz*2;

// ---------------- utility kernels ----------------
__global__ void zero_f32(float* __restrict__ p, int n) {
  int i = blockIdx.x * 256 + threadIdx.x;
  if (i < n) p[i] = 0.f;
}

// repack input (B,T,C,W) f32 -> [t][b][w][c] bf16 (channel-contiguous)
__global__ void conv_input(const float* __restrict__ in, __bf16* __restrict__ xin) {
  size_t idx = (size_t)blockIdx.x * 256 + threadIdx.x;
  if (idx >= (size_t)Bsz * Tsz * Csz * Wsz) return;
  int c = (int)(idx & 63);
  size_t rest = idx >> 6;
  int w = (int)(rest % 27); rest /= 27;
  int b = (int)(rest & 127);
  int t = (int)(rest >> 7);
  xin[idx] = (__bf16)in[(((size_t)b * Tsz + t) * Csz + c) * Wsz + w];
}

// pack conv weights -> bf16 panels B[n][k], k = tap*Cin + i (tap-major)
__global__ void prep_weights(const float* __restrict__ gw0, const float* __restrict__ cw0,
                             const float* __restrict__ aw0, const float* __restrict__ gw1,
                             const float* __restrict__ cw1, const float* __restrict__ aw1,
                             __bf16* __restrict__ wts) {
  int idx = blockIdx.x * 256 + threadIdx.x;
  if (idx >= (int)W_TOT) return;
  float v = 0.f;
  if (idx < (int)W_C0) {                        // gates0: [256][576], Cin=192
    int o = idx / KP_G0, k = idx % KP_G0;
    int tap = (k >= 192) + (k >= 384);
    int i = k - tap * 192;
    v = gw0[(o * 192 + i) * 3 + tap];
  } else if (idx < (int)W_G1) {                 // cand0: [128][608], Cin=200
    int r = idx - (int)W_C0;
    int o = r / KP_C0, k = r % KP_C0;
    if (k < 600) {
      int tap = (k >= 200) + (k >= 400);
      int i = k - tap * 200;
      v = (i < 192) ? cw0[(o * 192 + i) * 3 + tap]
                    : aw0[(o * 8 + (i - 192)) * 3 + tap];
    }
  } else if (idx < (int)W_C1) {                 // gates1: [256][768], Cin=256
    int r = idx - (int)W_G1;
    int o = r / KP_G1, k = r % KP_G1;
    int tap = (k >= 256) + (k >= 512);
    int i = k - tap * 256;
    v = gw1[(o * 256 + i) * 3 + tap];
  } else {                                      // cand1: [128][800], Cin=264
    int r = idx - (int)W_C1;
    int o = r / KP_C1, k = r % KP_C1;
    if (k < 792) {
      int tap = (k >= 264) + (k >= 528);
      int i = k - tap * 264;
      v = (i < 256) ? cw1[(o * 256 + i) * 3 + tap]
                    : aw1[(o * 8 + (i - 256)) * 3 + tap];
    }
  }
  wts[idx] = (__bf16)v;
}

// ---------------- fused GRU-step GEMM (WMMA bf16) ----------------
#define ASTRIDE 40   // 64 rows x 40 bf16 per buffer, 2 buffers (double-buffered)

// Branchless im2col gather of one 8-element group (32-bit offset math only;
// base pointer and offset selected with v_cndmask; zero page covers OOB).
__device__ __forceinline__ v8bf stage_load(
    int k, int sbw, int swid, int Ktot, int Cin,
    const __bf16* __restrict__ s0, int sc0, int cb0, int c0,
    const __bf16* __restrict__ s1, int sc1, int cb1, int c1,
    const __bf16* __restrict__ s2, int sc2, int cb2,
    const __bf16* __restrict__ zpage) {
  int tap = (k >= Cin) + (k >= 2 * Cin);
  int i   = k - tap * Cin;                 // 8 consecutive channels, one segment
  int wp  = swid + tap - 1;                // SAME padding
  int row = sbw + wp;                      // b*27 + wp
  int off0 = row * sc0 + cb0 + i;
  int off1 = row * sc1 + cb1 + (i - c0);
  int off2 = row * sc2 + cb2 + (i - c1);
  const __bf16* base = (i < c0) ? s0 : ((i < c1) ? s1 : s2);
  int off            = (i < c0) ? off0 : ((i < c1) ? off1 : off2);
  int ok = (int)(k < Ktot) & (int)((unsigned)wp < (unsigned)Wsz);
  const __bf16* p = ok ? (base + off) : zpage;
  return *(const v8bf*)p;
}

// MODE 0: gates -> sigmoid, write r*h (bf16) and u (f32)
// MODE 1: cand  -> + biases, mish((1-u)h + u*cnm), write pre (f32), channel stats
template <int MODE>
__global__ __launch_bounds__(256, 2)
void gru_gemm(const __bf16* __restrict__ s0, int sc0, int cb0, int c0,
              const __bf16* __restrict__ s1, int sc1, int cb1, int c1,
              const __bf16* __restrict__ s2, int sc2, int cb2, int Cin,
              const __bf16* __restrict__ Bmat, int Kpad,
              const float* __restrict__ bias0, const float* __restrict__ bias1,
              const __bf16* __restrict__ hbuf,
              __bf16* __restrict__ rh, float* __restrict__ ubuf,
              float* __restrict__ prebuf, float* __restrict__ stats,
              const __bf16* __restrict__ zpage) {
  __shared__ __bf16 As[2 * 64 * ASTRIDE];
  __shared__ float sS[64], sQ[64];

  const int tid  = threadIdx.x;
  const int lane = tid & 31, wv = tid >> 5;
  const int mi = wv & 3, ni = wv >> 2;       // 4 M-subtiles x 2 N-subtiles(32)
  const int l16 = lane & 15, hf = lane >> 4;

  if (MODE == 1 && tid < 64) { sS[tid] = 0.f; sQ[tid] = 0.f; }

  // staging coords: thread -> (row, 8-wide k group)
  const int srow = tid >> 2;
  const int skb  = (tid & 3) * 8;
  const int sm   = blockIdx.x * 64 + srow;   // < 3456 always
  const int sbch = sm / Wsz;
  const int swid = sm - sbch * Wsz;
  const int sbw  = sbch * Wsz;               // b*27
  const int Ktot = 3 * Cin;
  __bf16* const Aslot = As + srow * ASTRIDE + skb;

  v8f acc0 = {0.f,0.f,0.f,0.f,0.f,0.f,0.f,0.f};
  v8f acc1 = {0.f,0.f,0.f,0.f,0.f,0.f,0.f,0.f};

  // each lane owns B column n = tileN + ni*32 + l16, 16 contiguous K per frag
  const __bf16* Bcol = Bmat + (size_t)(blockIdx.y * 64 + ni * 32 + l16) * Kpad;

  // prologue: stage slice 0
  *(v8bf*)Aslot = stage_load(skb, sbw, swid, Ktot, Cin,
                             s0, sc0, cb0, c0, s1, sc1, cb1, c1, s2, sc2, cb2, zpage);
  __syncthreads();

  for (int kk = 0; kk < Kpad; kk += 32) {
    const int p = (kk >> 5) & 1;
    const bool more = (kk + 32 < Kpad);

    // issue next-slice staging load FIRST so its latency overlaps the body
    v8bf nval;
    if (more)
      nval = stage_load(kk + 32 + skb, sbw, swid, Ktot, Cin,
                        s0, sc0, cb0, c0, s1, sc1, cb1, c1, s2, sc2, cb2, zpage);

    const __bf16* ap = As + p * (64 * ASTRIDE) + (mi * 16 + l16) * ASTRIDE + hf * 16;
    v16bf afrag = cat8(*(const v8bf*)ap, *(const v8bf*)(ap + 8));

    const __bf16* bp0 = Bcol + kk + hf * 16;
    v16bf bf0 = cat8(*(const v8bf*)bp0, *(const v8bf*)(bp0 + 8));
    const __bf16* bp1 = bp0 + (size_t)16 * Kpad;
    v16bf bf1 = cat8(*(const v8bf*)bp1, *(const v8bf*)(bp1 + 8));

    if (more) {
      prefetch_wgp(bp0 + 32);                // WGP-scope global_prefetch_b8
      *(v8bf*)(Aslot + (1 - p) * (64 * ASTRIDE)) = nval;
    }
    __syncthreads();                         // single barrier per slice

    acc0 = wmma_bf16(afrag, bf0, acc0);
    acc1 = wmma_bf16(afrag, bf1, acc1);
  }

  // -------- epilogue (channel-last layout: ix = m*128 + col) --------
  const int colg0 = blockIdx.y * 64 + ni * 32 + l16;
  const int colg1 = colg0 + 16;
  const int mbase = blockIdx.x * 64 + mi * 16 + hf * 8;
  float S0 = 0.f, Q0 = 0.f, S1 = 0.f, Q1 = 0.f;

  const float bA0 = bias0[colg0];
  const float bA1 = bias0[colg1];
  const float bB0 = (MODE == 1) ? bias1[colg0] : 0.f;
  const float bB1 = (MODE == 1) ? bias1[colg1] : 0.f;

#pragma unroll
  for (int r = 0; r < 8; ++r) {
    const int m = mbase + r;                 // C layout: M = r + 8*hf
    if (MODE == 0) {
      float g0 = 1.f / (1.f + __expf(-(acc0[r] + bA0)));
      if (colg0 < Hsz) {
        int ix = m * Hsz + colg0;
        rh[ix] = (__bf16)(g0 * (float)hbuf[ix]);
      } else {
        ubuf[m * Hsz + (colg0 - Hsz)] = g0;
      }
      float g1 = 1.f / (1.f + __expf(-(acc1[r] + bA1)));
      if (colg1 < Hsz) {
        int ix = m * Hsz + colg1;
        rh[ix] = (__bf16)(g1 * (float)hbuf[ix]);
      } else {
        ubuf[m * Hsz + (colg1 - Hsz)] = g1;
      }
    } else {
      {
        int ix = m * Hsz + colg0;
        float uu = ubuf[ix], hh = (float)hbuf[ix];
        float z  = (1.f - uu) * hh + uu * (acc0[r] + bA0 + bB0);
        float sp = (z > 20.f) ? z : log1pf(__expf(z));
        float pr = z * tanhf(sp);            // mish
        prebuf[ix] = pr; S0 += pr; Q0 += pr * pr;
      }
      {
        int ix = m * Hsz + colg1;
        float uu = ubuf[ix], hh = (float)hbuf[ix];
        float z  = (1.f - uu) * hh + uu * (acc1[r] + bA1 + bB1);
        float sp = (z > 20.f) ? z : log1pf(__expf(z));
        float pr = z * tanhf(sp);
        prebuf[ix] = pr; S1 += pr; Q1 += pr * pr;
      }
    }
  }

  if (MODE == 1) {
    // ds_add_f32 LDS reduction, then one global atomic per channel per block
    atomicAdd(&sS[ni * 32 + l16], S0);
    atomicAdd(&sQ[ni * 32 + l16], Q0);
    atomicAdd(&sS[ni * 32 + l16 + 16], S1);
    atomicAdd(&sQ[ni * 32 + l16 + 16], Q1);
    __syncthreads();
    if (tid < 64) {
      atomicAdd(&stats[blockIdx.y * 64 + tid], sS[tid]);
      atomicAdd(&stats[Hsz + blockIdx.y * 64 + tid], sQ[tid]);
    }
  }
}

// batchnorm over (batch, width): h = g*(pre-mean)*rsqrt(var+eps)+b -> bf16
__global__ void bn_kernel(const float* __restrict__ pre, const float* __restrict__ stats,
                          const float* __restrict__ g, const float* __restrict__ bb,
                          __bf16* __restrict__ hout) {
  int idx = blockIdx.x * 256 + threadIdx.x;
  if (idx >= Bsz * HW) return;
  int ch = idx & (Hsz - 1);                  // channel-last
  const float invn = 1.f / (float)(Bsz * Wsz);
  float mean = stats[ch] * invn;
  float var  = stats[Hsz + ch] * invn - mean * mean;
  hout[idx] = (__bf16)(g[ch] * (pre[idx] - mean) * rsqrtf(var + 1e-5f) + bb[ch]);
}

// reward[b] = sum_{w,ch} h2[(b*27+w)*128+ch] * cw[ch*27+w] + cb
__global__ void reward_kernel(const __bf16* __restrict__ h2, const float* __restrict__ cw,
                              const float* __restrict__ cb, float* __restrict__ rout) {
  __shared__ float red[256];
  int b = blockIdx.x;
  float s = 0.f;
  for (int i = threadIdx.x; i < HW; i += 256) {
    int wi = i >> 7, ch = i & 127;
    s += (float)h2[(size_t)b * HW + i] * cw[ch * Wsz + wi];
  }
  red[threadIdx.x] = s;
  __syncthreads();
  for (int off = 128; off; off >>= 1) {
    if (threadIdx.x < off) red[threadIdx.x] += red[threadIdx.x + off];
    __syncthreads();
  }
  if (threadIdx.x == 0) rout[b] = red[0] + cb[0];
}

__global__ void final_kernel(const float* __restrict__ r, float* __restrict__ out) {
  int b = threadIdx.x;
  if (b < Bsz) out[b] = 0.99f * (r[Bsz + b] - r[b]);
}

// ---------------- host orchestration ----------------
extern "C" void kernel_launch(void* const* d_in, const int* in_sizes, int n_in,
                              void* d_out, int out_size, void* d_ws, size_t ws_size,
                              hipStream_t stream) {
  (void)in_sizes; (void)n_in; (void)out_size;
  if (ws_size < WS_NEED) return;

  const float* input = (const float*)d_in[0];
  const float* gw0 = (const float*)d_in[1];  const float* gb0 = (const float*)d_in[2];
  const float* cw0 = (const float*)d_in[3];  const float* cb0 = (const float*)d_in[4];
  const float* aw0 = (const float*)d_in[5];  const float* ab0 = (const float*)d_in[6];
  const float* bg0 = (const float*)d_in[7];  const float* bb0 = (const float*)d_in[8];
  const float* gw1 = (const float*)d_in[9];  const float* gb1 = (const float*)d_in[10];
  const float* cw1 = (const float*)d_in[11]; const float* cb1 = (const float*)d_in[12];
  const float* aw1 = (const float*)d_in[13]; const float* ab1 = (const float*)d_in[14];
  const float* bg1 = (const float*)d_in[15]; const float* bb1 = (const float*)d_in[16];
  const float* cnw = (const float*)d_in[17]; const float* cnb = (const float*)d_in[18];

  char* ws = (char*)d_ws;
  __bf16* h0   = (__bf16*)(ws + OFF_H0);
  __bf16* h1   = (__bf16*)(ws + OFF_H1);
  __bf16* rhb  = (__bf16*)(ws + OFF_RH);
  float*  ub   = (float*)(ws + OFF_U);
  float*  preb = (float*)(ws + OFF_PRE);
  float*  stA  = (float*)(ws + OFF_ST);       // [t][layer][256]
  float*  rwd  = (float*)(ws + OFF_RW);
  __bf16* zpg  = (__bf16*)(ws + OFF_ZP);
  __bf16* wts  = (__bf16*)(ws + OFF_WT);
  __bf16* xin  = (__bf16*)(ws + OFF_XIN);

  // one-time (per call) packing + state init
  const size_t nin = (size_t)Bsz * Tsz * Csz * Wsz;
  conv_input<<<(int)((nin + 255) / 256), 256, 0, stream>>>(input, xin);
  prep_weights<<<(int)((W_TOT + 255) / 256), 256, 0, stream>>>(gw0, cw0, aw0, gw1, cw1, aw1, wts);
  const int nact = Bsz * HW;
  zero_f32<<<(nact + 255) / 256, 256, 0, stream>>>((float*)h0, nact);   // h0+h1 bf16 contiguous
  // stats slab (all steps) + reward + zero page, contiguous region
  const int nzt = (int)((SZ_ST + 1024 + 64) / 4);
  zero_f32<<<(nzt + 255) / 256, 256, 0, stream>>>(stA, nzt);

  const dim3 gridG(Mtot / 64, 4), gridC(Mtot / 64, 2), blk(256);
  const int bnBlocks = (nact + 255) / 256;

  for (int t = 0; t < Tsz; ++t) {
    const __bf16* xt = xin + (size_t)t * Bsz * Wsz * Csz;   // [b][w][c] slab
    float* st0 = stA + (size_t)(t * 2 + 0) * 256;
    float* st1 = stA + (size_t)(t * 2 + 1) * 256;

    // ---- layer 0 ----
    gru_gemm<0><<<gridG, blk, 0, stream>>>(
        xt, 64, 0, 64,  h0, 128, 0, 192,  h0, 128, 0, 192,
        wts + W_G0, KP_G0, gb0, nullptr, h0, rhb, ub, preb, st0, zpg);
    gru_gemm<1><<<gridC, blk, 0, stream>>>(
        xt, 64, 0, 64,  rhb, 128, 0, 192,  xt, 64, 56, 200,
        wts + W_C0, KP_C0, cb0, ab0, h0, rhb, ub, preb, st0, zpg);
    bn_kernel<<<bnBlocks, blk, 0, stream>>>(preb, st0, bg0, bb0, h0);

    // ---- layer 1 (x = h0) ----
    gru_gemm<0><<<gridG, blk, 0, stream>>>(
        h0, 128, 0, 128,  h1, 128, 0, 256,  h1, 128, 0, 256,
        wts + W_G1, KP_G1, gb1, nullptr, h1, rhb, ub, preb, st1, zpg);
    gru_gemm<1><<<gridC, blk, 0, stream>>>(
        h0, 128, 0, 128,  rhb, 128, 0, 256,  xt, 64, 56, 264,
        wts + W_C1, KP_C1, cb1, ab1, h1, rhb, ub, preb, st1, zpg);
    bn_kernel<<<bnBlocks, blk, 0, stream>>>(preb, st1, bg1, bb1, h1);

    if (t == Tsz - 2) reward_kernel<<<Bsz, blk, 0, stream>>>(h1, cnw, cnb, rwd);
    if (t == Tsz - 1) reward_kernel<<<Bsz, blk, 0, stream>>>(h1, cnw, cnb, rwd + Bsz);
  }

  final_kernel<<<1, 128, 0, stream>>>(rwd, (float*)d_out);
}
